// GCN_GMT_62423054680393
// MI455X (gfx1250) — matrix operand order repaired and checked by reading, not compiled, not run on hardware
//
#include <hip/hip_runtime.h>
#include <hip/hip_bf16.h>
#include <math.h>
#include <stdint.h>

typedef __attribute__((ext_vector_type(16))) _Float16 v16h;
typedef __attribute__((ext_vector_type(8)))  float    v8f;
typedef int v4i __attribute__((vector_size(16)));

// ---------------- problem sizes (match reference) ----------------
static constexpr int Bg    = 256;        // graphs
static constexpr int NPG   = 300;        // nodes per graph
static constexpr int Nn    = Bg * NPG;   // 76800 nodes
static constexpr int EPG   = 3000;
static constexpr int Ee    = Bg * EPG;   // 768000 edges
static constexpr int HID   = 256;
static constexpr int OUTC  = 128;
static constexpr int K1    = 75;
static constexpr int HEADS = 8;

// ---------------- CDNA5 async global->LDS copy (ASYNCcnt-tracked) ----------------
#if defined(__has_builtin)
# if __has_builtin(__builtin_amdgcn_global_load_async_to_lds_b128)
#  define USE_ASYNC_LDS 1
# endif
#endif
#ifndef USE_ASYNC_LDS
# define USE_ASYNC_LDS 0
#endif

__device__ __forceinline__ void copy16_g2l(const float* g, float* l)
{
#if USE_ASYNC_LDS
    // per-lane async DMA of 16 bytes global -> LDS; tracked by ASYNCcnt
    __builtin_amdgcn_global_load_async_to_lds_b128(
        (__attribute__((address_space(1))) v4i*)(uintptr_t)g,
        (__attribute__((address_space(3))) v4i*)(unsigned int)(uintptr_t)l,
        0, 0);
#else
    *(float4*)l = *(const float4*)g;
#endif
}

__device__ __forceinline__ void wait_async_lds()
{
#if USE_ASYNC_LDS
# if __has_builtin(__builtin_amdgcn_s_wait_asynccnt)
    __builtin_amdgcn_s_wait_asynccnt(0);
# else
    asm volatile("s_wait_asynccnt 0x0" ::: "memory");
# endif
#endif
}

// ---------------- WMMA GEMM: C[M,N] = A[M,KC] * W[KC,N] (+bias) ----------------
// Double-buffered LDS (f32 tiles staged by async DMA), f16 fragments, f32 accum.
// Block tile 64x64, 8 waves, each wave a 16x32 tile (2 WMMAs per K-step).
template<int KC>
__global__ __launch_bounds__(256) void gemm_wmma(const float* __restrict__ A,
                                                 const float* __restrict__ W,
                                                 const float* __restrict__ bias,
                                                 float* __restrict__ C,
                                                 int M, int N)
{
    constexpr int S = KC / 32;
    __shared__ __align__(16) float Asf[2][64 * 32];   // [row][k]
    __shared__ __align__(16) float Wsf[2][32 * 64];   // [k][col]
    const int tid  = threadIdx.x;
    const int m0   = blockIdx.x * 64;
    const int n0   = blockIdx.y * 64;
    const int lane = tid & 31;         // wave32
    const int wid  = tid >> 5;         // 0..7
    const int wm   = wid >> 1;         // 0..3 (M sub-tile)
    const int wn   = wid & 1;          // 0..1 (N sub-tile)
    const int lr   = lane & 15;
    const int kh   = lane >> 4;        // K-half held by this lane

    // staging: A tile = 512 16B chunks, W tile = 512 16B chunks; 2 each per thread
    const int ca0 = tid, ca1 = tid + 256;
    const int ar0 = ca0 >> 3, aq0 = (ca0 & 7) * 4;
    const int ar1 = ca1 >> 3, aq1 = (ca1 & 7) * 4;
    const float* gA0 = A + (size_t)(m0 + ar0) * KC + aq0;
    const float* gA1 = A + (size_t)(m0 + ar1) * KC + aq1;
    const int la0 = ar0 * 32 + aq0;
    const int la1 = ar1 * 32 + aq1;

    const int wk0 = ca0 >> 4, wq0 = (ca0 & 15) * 4;
    const int wk1 = ca1 >> 4, wq1 = (ca1 & 15) * 4;
    const float* gW0 = W + (size_t)wk0 * N + n0 + wq0;
    const float* gW1 = W + (size_t)wk1 * N + n0 + wq1;
    const int lw0 = wk0 * 64 + wq0;
    const int lw1 = wk1 * 64 + wq1;

    auto issue = [&](int ks) {
        const int buf = ks & 1;
        const int kk  = ks * 32;
        copy16_g2l(gA0 + kk, &Asf[buf][la0]);
        copy16_g2l(gA1 + kk, &Asf[buf][la1]);
        copy16_g2l(gW0 + (size_t)kk * N, &Wsf[buf][lw0]);
        copy16_g2l(gW1 + (size_t)kk * N, &Wsf[buf][lw1]);
    };

    v8f acc0 = {};
    v8f acc1 = {};

    issue(0);
#pragma unroll
    for (int ks = 0; ks < S; ++ks) {
        const int buf = ks & 1;
        wait_async_lds();            // this wave's stage-ks copies have landed
        __syncthreads();             // everyone's copies have landed
        if (ks + 1 < S) issue(ks + 1);   // prefetch next tile into other buffer

        // A fragment (16x32 f16): lane lr = row, kh selects K-half (contiguous k = 16*kh + s, same order as B)
        const int   row = wm * 16 + lr;
        const float* ap = &Asf[buf][row * 32 + 16 * kh];
        float4 f0 = *(const float4*)(ap);
        float4 f1 = *(const float4*)(ap + 4);
        float4 f2 = *(const float4*)(ap + 8);
        float4 f3 = *(const float4*)(ap + 12);
        v16h a;
        a[0]  = (_Float16)f0.x; a[1]  = (_Float16)f0.y; a[2]  = (_Float16)f0.z; a[3]  = (_Float16)f0.w;
        a[4]  = (_Float16)f1.x; a[5]  = (_Float16)f1.y; a[6]  = (_Float16)f1.z; a[7]  = (_Float16)f1.w;
        a[8]  = (_Float16)f2.x; a[9]  = (_Float16)f2.y; a[10] = (_Float16)f2.z; a[11] = (_Float16)f2.w;
        a[12] = (_Float16)f3.x; a[13] = (_Float16)f3.y; a[14] = (_Float16)f3.z; a[15] = (_Float16)f3.w;

        {
            int col = wn * 32 + lr;
            v16h b;
#pragma unroll
            for (int j = 0; j < 8; ++j) {
                b[2 * j]     = (_Float16)Wsf[buf][(kh * 16 + 2 * j) * 64 + col];
                b[2 * j + 1] = (_Float16)Wsf[buf][(kh * 16 + 2 * j + 1) * 64 + col];
            }
            acc0 = __builtin_amdgcn_wmma_f32_16x16x32_f16(false, a, false, b,
                                                          (short)0, acc0, false, false);
        }
        {
            int col = wn * 32 + 16 + lr;
            v16h b;
#pragma unroll
            for (int j = 0; j < 8; ++j) {
                b[2 * j]     = (_Float16)Wsf[buf][(kh * 16 + 2 * j) * 64 + col];
                b[2 * j + 1] = (_Float16)Wsf[buf][(kh * 16 + 2 * j + 1) * 64 + col];
            }
            acc1 = __builtin_amdgcn_wmma_f32_16x16x32_f16(false, a, false, b,
                                                          (short)0, acc1, false, false);
        }
        __syncthreads();             // done reading buf before it is re-filled
    }

    // store D (C/D layout: VGPR r -> row r (+8 for lanes 16..31), lane lr -> col)
#pragma unroll
    for (int t = 0; t < 2; ++t) {
        const v8f accv = t ? acc1 : acc0;
        int colg = n0 + wn * 32 + t * 16 + lr;
        float bv = bias ? bias[colg] : 0.f;
#pragma unroll
        for (int r = 0; r < 8; ++r) {
            int rowg = m0 + wm * 16 + 8 * kh + r;
            C[(size_t)rowg * N + colg] = accv[r] + bv;
        }
    }
}

// ---------------- elementwise / reduction helpers ----------------
__global__ void copy_kernel(float* __restrict__ dst, const float* __restrict__ src, int total)
{
    int i = blockIdx.x * blockDim.x + threadIdx.x;
    if (i < total) dst[i] = src[i];
}

__global__ void fill_kernel(float* __restrict__ dst, float v, int total)
{
    int i = blockIdx.x * blockDim.x + threadIdx.x;
    if (i < total) dst[i] = v;
}

// agg[dst[e], c] += x[src[e], c]
__global__ void edge_add_kernel(float* __restrict__ out, const float* __restrict__ x,
                                const int* __restrict__ src, const int* __restrict__ dst,
                                int C, int total)
{
    int idx = blockIdx.x * blockDim.x + threadIdx.x;
    if (idx >= total) return;
    int e = idx / C, c = idx - e * C;
    atomicAdd(&out[(size_t)dst[e] * C + c], x[(size_t)src[e] * C + c]);
}

__global__ void deg_edge_kernel(float* __restrict__ deg, const int* __restrict__ dst, int total)
{
    int e = blockIdx.x * blockDim.x + threadIdx.x;
    if (e < total) atomicAdd(&deg[dst[e]], 1.0f);
}

__global__ void dinv_kernel(float* __restrict__ dinv, const float* __restrict__ deg, int total)
{
    int i = blockIdx.x * blockDim.x + threadIdx.x;
    if (i < total) dinv[i] = rsqrtf(deg[i]);
}

// out[i,c] = dinv[i]^2 * xw[i,c] + bias[c]
__global__ void gcn_self_kernel(float* __restrict__ out, const float* __restrict__ xw,
                                const float* __restrict__ dinv, const float* __restrict__ bias,
                                int total)
{
    int idx = blockIdx.x * blockDim.x + threadIdx.x;
    if (idx >= total) return;
    int i = idx >> 8, c = idx & 255;
    float di = dinv[i];
    out[idx] = di * di * xw[idx] + bias[c];
}

// out[dst,c] += dinv[src]*dinv[dst]*xw[src,c]
__global__ void gcn_edge_kernel(float* __restrict__ out, const float* __restrict__ xw,
                                const float* __restrict__ dinv,
                                const int* __restrict__ src, const int* __restrict__ dst,
                                int total)
{
    int idx = blockIdx.x * blockDim.x + threadIdx.x;
    if (idx >= total) return;
    int e = idx >> 8, c = idx & 255;
    int s = src[e], d = dst[e];
    atomicAdd(&out[(size_t)d * 256 + c], dinv[s] * dinv[d] * xw[(size_t)s * 256 + c]);
}

// column batchnorm stats over M rows: stats[c]=mean, stats[Nc+c]=var
__global__ __launch_bounds__(256) void bn_stats_kernel(const float* __restrict__ Y,
                                                       float* __restrict__ stats, int M, int Nc)
{
    __shared__ float rs[256];
    __shared__ float rs2[256];
    int c = blockIdx.x, t = threadIdx.x;
    float s = 0.f, s2 = 0.f;
    for (int r = t; r < M; r += 256) {
        float v = Y[(size_t)r * Nc + c];
        s += v; s2 += v * v;
    }
    rs[t] = s; rs2[t] = s2;
    __syncthreads();
    for (int off = 128; off > 0; off >>= 1) {
        if (t < off) { rs[t] += rs[t + off]; rs2[t] += rs2[t + off]; }
        __syncthreads();
    }
    if (t == 0) {
        float mean = rs[0] / (float)M;
        stats[c]      = mean;
        stats[Nc + c] = rs2[0] / (float)M - mean * mean;
    }
}

__global__ void bn_relu_kernel(const float* __restrict__ Y, const float* __restrict__ stats,
                               const float* __restrict__ g, const float* __restrict__ be,
                               float* __restrict__ H, int total, int Nc)
{
    int idx = blockIdx.x * blockDim.x + threadIdx.x;
    if (idx >= total) return;
    int c = idx % Nc;
    float h = (Y[idx] - stats[c]) * rsqrtf(stats[Nc + c] + 1e-5f) * g[c] + be[c];
    H[idx] = fmaxf(h, 0.f);
}

// dst[r,:] = LayerNorm( A[(r%a_rpb or r), :] + (relu?)(Bm[r,:]) ) * g + be ; C=256
__global__ __launch_bounds__(256) void ln_kernel(const float* __restrict__ A, int a_rpb,
                                                 const float* __restrict__ Bm, int brelu,
                                                 const float* __restrict__ g,
                                                 const float* __restrict__ be,
                                                 float* __restrict__ dst)
{
    __shared__ float red[256];
    int r = blockIdx.x, c = threadIdx.x;
    size_t arow = a_rpb ? (size_t)(r % a_rpb) * 256 : (size_t)r * 256;
    float bv = Bm[(size_t)r * 256 + c];
    if (brelu) bv = fmaxf(bv, 0.f);
    float x = A[arow + c] + bv;
    red[c] = x;
    __syncthreads();
    for (int off = 128; off > 0; off >>= 1) {
        if (c < off) red[c] += red[c + off];
        __syncthreads();
    }
    float mean = red[0] * (1.f / 256.f);
    __syncthreads();
    float d = x - mean;
    red[c] = d * d;
    __syncthreads();
    for (int off = 128; off > 0; off >>= 1) {
        if (c < off) red[c] += red[c + off];
        __syncthreads();
    }
    float var = red[0] * (1.f / 256.f);
    dst[(size_t)r * 256 + c] = d * rsqrtf(var + 1e-5f) * g[c] + be[c];
}

// small row-vector GEMM: out[q,c] = sum_k S[q,k]*Wm[k,c] + bias[c], K=N=256
__global__ void rowmat_kernel(const float* __restrict__ S, const float* __restrict__ Wm,
                              const float* __restrict__ bias, float* __restrict__ out, int rows)
{
    int idx = blockIdx.x * blockDim.x + threadIdx.x;
    if (idx >= rows * 256) return;
    int q = idx >> 8, c = idx & 255;
    float s = bias[c];
    for (int k = 0; k < 256; ++k)
        s += S[q * 256 + k] * Wm[k * 256 + c];
    out[idx] = s;
}

// softmax attention, one block per (b,q,h). Q row either broadcast (q_bstride==0)
// or per-batch (q_bstride = qn*256 elems). K/V/O are [B, kn|qn, 256], head h at cols h*32.
__global__ __launch_bounds__(128) void attn_kernel(const float* __restrict__ Q,
                                                   const float* __restrict__ K,
                                                   const float* __restrict__ V,
                                                   float* __restrict__ O,
                                                   int qn, int kn, int q_bstride, float scale)
{
    __shared__ float qs[32];
    __shared__ float s[304];
    __shared__ float red[128];
    __shared__ float acc[32];
    int blk = blockIdx.x;
    int h   = blk & 7;
    int q   = (blk >> 3) % qn;
    int b   = blk / (8 * qn);
    const float* Qrow = Q + (size_t)b * q_bstride + (size_t)q * 256 + h * 32;
    const float* Kb   = K + (size_t)b * kn * 256 + h * 32;
    const float* Vb   = V + (size_t)b * kn * 256 + h * 32;
    int t = threadIdx.x;
    if (t < 32) { qs[t] = Qrow[t]; acc[t] = 0.f; }
    __syncthreads();

    float mx = -1e30f;
    for (int k = t; k < kn; k += 128) {
        const float* kr = Kb + (size_t)k * 256;
        float d = 0.f;
#pragma unroll
        for (int i = 0; i < 32; ++i) d += qs[i] * kr[i];
        d *= scale;
        s[k] = d;
        mx = fmaxf(mx, d);
    }
    red[t] = mx;
    __syncthreads();
    for (int off = 64; off > 0; off >>= 1) {
        if (t < off) red[t] = fmaxf(red[t], red[t + off]);
        __syncthreads();
    }
    float gmax = red[0];
    __syncthreads();

    float psum = 0.f;
    for (int k = t; k < kn; k += 128) {
        float e = __expf(s[k] - gmax);
        s[k] = e;
        psum += e;
    }
    red[t] = psum;
    __syncthreads();
    for (int off = 64; off > 0; off >>= 1) {
        if (t < off) red[t] += red[t + off];
        __syncthreads();
    }
    float inv = 1.f / red[0];

    float racc[32];
#pragma unroll
    for (int i = 0; i < 32; ++i) racc[i] = 0.f;
    for (int k = t; k < kn; k += 128) {
        float w = s[k];
        const float* vr = Vb + (size_t)k * 256;
#pragma unroll
        for (int i = 0; i < 32; ++i) racc[i] += w * vr[i];
    }
#pragma unroll
    for (int i = 0; i < 32; ++i) atomicAdd(&acc[i], racc[i]);   // ds_add_f32
    __syncthreads();
    if (t < 32)
        O[(size_t)b * qn * 256 + (size_t)q * 256 + h * 32 + t] = acc[t] * inv;
}

// ---------------- host-side orchestration ----------------
extern "C" void kernel_launch(void* const* d_in, const int* in_sizes, int n_in,
                              void* d_out, int out_size, void* d_ws, size_t ws_size,
                              hipStream_t stream)
{
    (void)in_sizes; (void)n_in; (void)out_size; (void)ws_size;
    // inputs: x, edge_index, batch, then params in dict order
    const float* x   = (const float*)d_in[0];
    const int*   ei  = (const int*)d_in[1];
    const int*   src = ei;
    const int*   dst = ei + Ee;
    auto P = [&](int i) -> const float* { return (const float*)d_in[i]; };
    // param indices (dict insertion order in setup_inputs)
    // 3..10: gin0 {w1,b1,g1,be1,w2,b2,g2,be2}; 11..18: gin1 {...}
    // 19 lin1w 20 lin1b 21 lin2w 22 lin2b 23 S1 24 S3
    // 25..36 p1_{fqw,fqb,kw,kb,vw,vb,fow,fob,ln0g,ln0b,ln1g,ln1b}; 37..48 p2_; 49..60 p3_

    const size_t NB = (size_t)Nn * 256;
    float* ws   = (float*)d_ws;
    float* B0   = ws;
    float* B1   = ws + NB;
    float* B2   = ws + 2 * NB;
    float* B3   = ws + 3 * NB;
    float* deg  = ws + 4 * NB;
    float* dinv = deg + Nn;
    float* stats= dinv + Nn;           // 512
    float* qr1  = stats + 512;         // 75*256
    float* q3   = qr1 + K1 * 256;      // 256

    const int TPB = 256;
    const int totN256 = Nn * 256;
    const int totN128 = Nn * 128;
    const int totE256 = Ee * 256;
    const int totE128 = Ee * 128;
    const dim3 gBig(Nn / 64, 256 / 64);       // M=76800, N=256
    const float scale = 0.0625f;              // 1/sqrt(256)

    // ---------- GIN layer 0 ----------
    copy_kernel<<<(totN128 + TPB - 1) / TPB, TPB, 0, stream>>>(B0, x, totN128);
    edge_add_kernel<<<(totE128 + TPB - 1) / TPB, TPB, 0, stream>>>(B0, x, src, dst, 128, totE128);
    gemm_wmma<128><<<gBig, 256, 0, stream>>>(B0, P(3), P(4), B1, Nn, 256);
    bn_stats_kernel<<<256, 256, 0, stream>>>(B1, stats, Nn, 256);
    bn_relu_kernel<<<(totN256 + TPB - 1) / TPB, TPB, 0, stream>>>(B1, stats, P(5), P(6), B0, totN256, 256);
    gemm_wmma<256><<<gBig, 256, 0, stream>>>(B0, P(7), P(8), B1, Nn, 256);
    bn_stats_kernel<<<256, 256, 0, stream>>>(B1, stats, Nn, 256);
    bn_relu_kernel<<<(totN256 + TPB - 1) / TPB, TPB, 0, stream>>>(B1, stats, P(9), P(10), B0, totN256, 256);

    // ---------- GIN layer 1 ----------
    copy_kernel<<<(totN256 + TPB - 1) / TPB, TPB, 0, stream>>>(B1, B0, totN256);
    edge_add_kernel<<<(totE256 + TPB - 1) / TPB, TPB, 0, stream>>>(B1, B0, src, dst, 256, totE256);
    gemm_wmma<256><<<gBig, 256, 0, stream>>>(B1, P(11), P(12), B2, Nn, 256);
    bn_stats_kernel<<<256, 256, 0, stream>>>(B2, stats, Nn, 256);
    bn_relu_kernel<<<(totN256 + TPB - 1) / TPB, TPB, 0, stream>>>(B2, stats, P(13), P(14), B1, totN256, 256);
    gemm_wmma<256><<<gBig, 256, 0, stream>>>(B1, P(15), P(16), B2, Nn, 256);
    bn_stats_kernel<<<256, 256, 0, stream>>>(B2, stats, Nn, 256);
    bn_relu_kernel<<<(totN256 + TPB - 1) / TPB, TPB, 0, stream>>>(B2, stats, P(17), P(18), B1, totN256, 256);

    // ---------- lin1 ----------
    gemm_wmma<256><<<gBig, 256, 0, stream>>>(B1, P(19), P(20), B0, Nn, 256);   // xh -> B0

    // ---------- degrees (dst + self loop) ----------
    fill_kernel<<<(Nn + TPB - 1) / TPB, TPB, 0, stream>>>(deg, 1.0f, Nn);
    deg_edge_kernel<<<(Ee + TPB - 1) / TPB, TPB, 0, stream>>>(deg, dst, Ee);
    dinv_kernel<<<(Nn + TPB - 1) / TPB, TPB, 0, stream>>>(dinv, deg, Nn);

    // ---------- pool1: GCN keys/values ----------
    gemm_wmma<256><<<gBig, 256, 0, stream>>>(B0, P(27), nullptr, B1, Nn, 256); // xk
    gcn_self_kernel<<<(totN256 + TPB - 1) / TPB, TPB, 0, stream>>>(B2, B1, dinv, P(28), totN256);
    gcn_edge_kernel<<<(totE256 + TPB - 1) / TPB, TPB, 0, stream>>>(B2, B1, dinv, src, dst, totE256); // Kg -> B2
    gemm_wmma<256><<<gBig, 256, 0, stream>>>(B0, P(29), nullptr, B1, Nn, 256); // xv
    gcn_self_kernel<<<(totN256 + TPB - 1) / TPB, TPB, 0, stream>>>(B0, B1, dinv, P(30), totN256);
    gcn_edge_kernel<<<(totE256 + TPB - 1) / TPB, TPB, 0, stream>>>(B0, B1, dinv, src, dst, totE256); // Vg -> B0

    // ---------- pool1: PMA (75 seeds, keys = 300 nodes) ----------
    rowmat_kernel<<<(K1 * 256 + TPB - 1) / TPB, TPB, 0, stream>>>(P(23), P(25), P(26), qr1, K1);
    attn_kernel<<<Bg * K1 * HEADS, 128, 0, stream>>>(qr1, B2, B0, B1, K1, NPG, 0, scale); // O -> B1
    ln_kernel<<<Bg * K1, 256, 0, stream>>>(qr1, K1, B1, 0, P(33), P(34), B2);  // t -> B2
    gemm_wmma<256><<<dim3(Bg * K1 / 64, 4), 256, 0, stream>>>(B2, P(31), P(32), B0, Bg * K1, 256);
    ln_kernel<<<Bg * K1, 256, 0, stream>>>(B2, 0, B0, 1, P(35), P(36), B1);    // bx -> B1

    // ---------- pool2: SAB (75x75) ----------
    gemm_wmma<256><<<dim3(Bg * K1 / 64, 4), 256, 0, stream>>>(B1, P(37), P(38), B0, Bg * K1, 256); // Q
    gemm_wmma<256><<<dim3(Bg * K1 / 64, 4), 256, 0, stream>>>(B1, P(39), P(40), B2, Bg * K1, 256); // K
    gemm_wmma<256><<<dim3(Bg * K1 / 64, 4), 256, 0, stream>>>(B1, P(41), P(42), B3, Bg * K1, 256); // V
    attn_kernel<<<Bg * K1 * HEADS, 128, 0, stream>>>(B0, B2, B3, B1, K1, K1, K1 * 256, scale);     // O -> B1
    ln_kernel<<<Bg * K1, 256, 0, stream>>>(B0, 0, B1, 0, P(45), P(46), B2);
    gemm_wmma<256><<<dim3(Bg * K1 / 64, 4), 256, 0, stream>>>(B2, P(43), P(44), B3, Bg * K1, 256);
    ln_kernel<<<Bg * K1, 256, 0, stream>>>(B2, 0, B3, 1, P(47), P(48), B1);    // bx2 -> B1

    // ---------- pool3: PMA with 1 seed ----------
    rowmat_kernel<<<1, 256, 0, stream>>>(P(24), P(49), P(50), q3, 1);
    gemm_wmma<256><<<dim3(Bg * K1 / 64, 4), 256, 0, stream>>>(B1, P(51), P(52), B0, Bg * K1, 256); // K3
    gemm_wmma<256><<<dim3(Bg * K1 / 64, 4), 256, 0, stream>>>(B1, P(53), P(54), B2, Bg * K1, 256); // V3
    attn_kernel<<<Bg * 1 * HEADS, 128, 0, stream>>>(q3, B0, B2, B3, 1, K1, 0, scale);              // O3 -> B3 [256,256]
    ln_kernel<<<Bg, 256, 0, stream>>>(q3, 1, B3, 0, P(57), P(58), B0);
    gemm_wmma<256><<<dim3(Bg / 64, 4), 256, 0, stream>>>(B0, P(55), P(56), B2, Bg, 256);
    ln_kernel<<<Bg, 256, 0, stream>>>(B0, 0, B2, 1, P(59), P(60), B3);

    // ---------- final linear [256,256] @ [256,128] ----------
    gemm_wmma<256><<<dim3(Bg / 64, OUTC / 64), 256, 0, stream>>>(B3, P(21), P(22), (float*)d_out, Bg, OUTC);
}
